// Net_LSV_21148418966327
// MI455X (gfx1250) — compile-verified, matrix operand order checked by reading, not run on hardware
//
#include <hip/hip_runtime.h>
#include <hip/hip_bf16.h>

typedef __attribute__((ext_vector_type(16))) _Float16 v16h;
typedef __attribute__((ext_vector_type(8)))  float    v8f;

#define KTMAX 4
#define PPB   128            // paths per block
#define NSTEP 96
#define PL    24
#define MC    131072

// ---------------- ws layout (bytes) ----------------
// [0,2048)            : 512 f32 stats accumulators
// [2048, +274432)     : svol fragments, 4 mats x 34304 halves
// [276480, +57344)    : hvv  fragments, 4 mats x 7168 halves
// [333824, +24576)    : cv   fragments, 4 mats x 3072 halves
#define WS_SV  2048
#define WS_HV  276480
#define WS_CV  333824

// fragment half-offsets inside one maturity blob
#define SV_L1 0
#define SV_L2 3584
#define SV_L3 17920
#define SV_L4 32256
#define SV_MAT 34304
#define HV_L1 0
#define HV_L2 2048
#define HV_L3 6144
#define HV_MAT 7168
#define CV_L1 0
#define CV_L2 1024
#define CV_L3 2048
#define CV_MAT 3072

__device__ __forceinline__ float sp_f(float x) {           // softplus
  return (x > 20.f) ? x : log1pf(expf(x));
}
__device__ __forceinline__ float wred(float x) {           // wave32 sum
  #pragma unroll
  for (int o = 16; o > 0; o >>= 1) x += __shfl_xor(x, o, 32);
  return x;
}

// ---------------- weight pre-swizzle kernels ----------------
// fragment element (tile, lane, half): k = kt*32 + koff(lane,half), n = nt*16 + (lane&15)
__global__ void k_pack(const float* __restrict__ W, int wMatStride,
                       _Float16* __restrict__ dst, int dMatStride,
                       int Kv, int Nv, int KT, int NT) {
  int m = blockIdx.y;
  int t = blockIdx.x * blockDim.x + threadIdx.x;
  int total = KT * NT * 512;
  if (t >= total) return;
  int half = t & 15, lane = (t >> 4) & 31, tile = t >> 9;
  int nt = tile % NT, kt = tile / NT;
  int k = kt * 32 + ((half & 7) | ((half & 8) << 1) | ((lane & 16) >> 1));
  int n = nt * 16 + (lane & 15);
  float v = (k < Kv && n < Nv) ? W[(size_t)m * wMatStride + k * Nv + n] : 0.f;
  dst[(size_t)m * dMatStride + tile * 512 + lane * 16 + half] = (_Float16)v;
}

struct Blk { const float* W; int ms, k0, k1, n0, n1, ld; };

__global__ void k_pack3(Blk a, Blk b, Blk c,
                        _Float16* __restrict__ dst, int dMatStride, int KT, int NT) {
  int m = blockIdx.y;
  int t = blockIdx.x * blockDim.x + threadIdx.x;
  int total = KT * NT * 512;
  if (t >= total) return;
  int half = t & 15, lane = (t >> 4) & 31, tile = t >> 9;
  int nt = tile % NT, kt = tile / NT;
  int k = kt * 32 + ((half & 7) | ((half & 8) << 1) | ((lane & 16) >> 1));
  int n = nt * 16 + (lane & 15);
  float v = 0.f;
  if      (k >= a.k0 && k < a.k1 && n >= a.n0 && n < a.n1) v = a.W[(size_t)m*a.ms + (k-a.k0)*a.ld + (n-a.n0)];
  else if (k >= b.k0 && k < b.k1 && n >= b.n0 && n < b.n1) v = b.W[(size_t)m*b.ms + (k-b.k0)*b.ld + (n-b.n0)];
  else if (k >= c.k0 && k < c.k1 && n >= c.n0 && n < c.n1) v = c.W[(size_t)m*c.ms + (k-c.k0)*c.ld + (n-c.n0)];
  dst[(size_t)m * dMatStride + tile * 512 + lane * 16 + half] = (_Float16)v;
}

// ---------------- main simulation kernel ----------------
// MODE 0: relu -> f16 scatter into next A-fragment buffer
// MODE 1: raw f32 column outputs -> outArr[n*PPB + pw + P]   (n < NC)
// MODE 2: softplus * dS -> accumulate into cv_van LDS (mat m)
template<int KT, int NT, int MODE>
__device__ __forceinline__ void gemm_layer(
    const _Float16* __restrict__ Wt, const float* __restrict__ bias,
    const _Float16* __restrict__ bufIn, _Float16* __restrict__ bufOut,
    float* outArr, int NC, float* cvv, const float* dSl, int m,
    int lane, int pw)
{
  #pragma unroll
  for (int mt = 0; mt < 2; ++mt) {
    v16h a[KT];
    #pragma unroll
    for (int kt = 0; kt < KT; ++kt)
      a[kt] = *(const v16h*)(bufIn + ((mt * KTMAX + kt) << 9) + lane * 16);
    #pragma unroll
    for (int nt = 0; nt < NT; ++nt) {
      v8f cacc = {};
      #pragma unroll
      for (int kt = 0; kt < KT; ++kt) {
        v16h b = *(const v16h*)(Wt + ((kt * NT + nt) << 9) + lane * 16);
        cacc = __builtin_amdgcn_wmma_f32_16x16x32_f16(false, a[kt], false, b,
                                                      (short)0, cacc, false, false);
      }
      int n = nt * 16 + (lane & 15);
      float bv = bias[n];
      int Pbase = mt * 16 + ((lane >> 4) << 3);
      #pragma unroll
      for (int r = 0; r < 8; ++r) {
        float y = cacc[r] + bv;
        int P = Pbase + r;
        if (MODE == 0) {
          y = fmaxf(y, 0.f);
          int kk = n & 31, kt2 = n >> 5;
          int la = (P & 15) | (((kk >> 3) & 1) << 4);
          int j  = (kk & 7) | ((kk >> 4) << 3);
          bufOut[(((P >> 4) * KTMAX + kt2) << 9) + la * 16 + j] = (_Float16)y;
        } else if (MODE == 1) {
          if (n < NC) outArr[n * PPB + pw + P] = y;
        } else {
          if (n < 21) cvv[(pw + P) * 84 + m * 21 + n] += sp_f(y) * dSl[pw + P];
        }
      }
    }
  }
}

__device__ __forceinline__ void zero_buf(_Float16* buf, int lane) {
  v16h zv = {};
  #pragma unroll
  for (int q = 0; q < 8; ++q)
    *(v16h*)(buf + ((q * 32 + lane) << 4)) = zv;
}

// each lane writes only its own fragment slice (kt=0), data pulled by shfl
__device__ __forceinline__ void build_A0(_Float16* buf, int lane,
                                         float t0, float Slog, float V) {
  #pragma unroll
  for (int mt = 0; mt < 2; ++mt) {
    int srcl = (lane & 15) | (mt << 4);
    float sl = __shfl(Slog, srcl, 32);
    float vv = __shfl(V, srcl, 32);
    v16h hv = {};
    if (lane < 16) { hv[0] = (_Float16)t0; hv[1] = (_Float16)sl; hv[2] = (_Float16)vv; }
    *(v16h*)(buf + ((mt * KTMAX) << 9) + lane * 16) = hv;
  }
}

__global__ __launch_bounds__(128, 1)
void k_sim(const float* __restrict__ S0p, const float* __restrict__ ratep,
           const float* __restrict__ z,  const float* __restrict__ zz,
           const float* __restrict__ tg, const float* __restrict__ strikes,
           const float* svb0, const float* svb1, const float* svb2, const float* svb3,
           const float* cvb0, const float* cvb1, const float* cvb2,
           const float* hb0,  const float* hb1,  const float* hb2,
           const float* db0,  const float* db1,  const float* db2,
           const float* vb0,  const float* vb1,  const float* vb2,
           const float* v0p,  const float* rhop,
           const _Float16* __restrict__ wsv, const _Float16* __restrict__ whv,
           const _Float16* __restrict__ wcv, float* __restrict__ stats)
{
  __shared__ __align__(32) _Float16 sWsv[34304];
  __shared__ __align__(32) _Float16 sWhv[7168];
  __shared__ __align__(32) _Float16 sWcv[12288];
  __shared__ __align__(32) _Float16 sAct[4 * 2 * 4096];   // per wave: 2 ping-pong buffers
  __shared__ float sCvv[PPB * 84];
  __shared__ float sBsv[352];     // b0@0(112) b1@112 b2@224 b3@336(16)
  __shared__ float sBhv[144];     // b0@0(64) b1@64 b2@128(16)
  __shared__ float sBcv[384];     // 4 mats x (b0@0(32) b1@32 b2@64(32))
  __shared__ float sPd[PPB];
  __shared__ float sHvv[3 * PPB];
  __shared__ float sDS[PPB];

  const int tid  = threadIdx.x;
  const int wave = tid >> 5, lane = tid & 31;
  const int pw   = wave * 32;
  const int pl   = tid;                       // wave-local ownership: lane <-> path
  const size_t pg = (size_t)blockIdx.x * PPB + tid;

  const float S0    = S0p[0];
  const float rate  = ratep[0];
  const float rho_t = tanhf(rhop[0]);
  float Slog = logf(S0);
  float V    = 0.5f * (1.f / (1.f + expf(-v0p[0])));
  float run_max = S0;
  float cv_ex = 0.f;

  for (int j = tid; j < PPB * 84; j += PPB) sCvv[j] = 0.f;

  _Float16* bufA = sAct + wave * 8192;
  _Float16* bufB = bufA + 4096;

  #pragma clang loop unroll(disable)
  for (int i = 1; i <= NSTEP; ++i) {
    const int idx = (i - 1) / PL;
    if (((i - 1) % PL) == 0) {                 // stage this maturity phase's weights
      __syncthreads();
      { const uint4* s = (const uint4*)(wsv + idx * SV_MAT); uint4* d = (uint4*)sWsv;
        for (int q = tid; q < 34304 / 8; q += PPB) d[q] = s[q]; }
      { const uint4* s = (const uint4*)(whv + idx * HV_MAT); uint4* d = (uint4*)sWhv;
        for (int q = tid; q < 7168 / 8; q += PPB) d[q] = s[q]; }
      if (i == 1) {
        const uint4* s = (const uint4*)wcv; uint4* d = (uint4*)sWcv;
        for (int q = tid; q < 12288 / 8; q += PPB) d[q] = s[q];
        for (int m = 0; m < 4; ++m)
          for (int j = tid; j < 96; j += PPB) {
            float v;
            if (j < 32)      v = (j < 20)      ? cvb0[m * 20 + j]        : 0.f;
            else if (j < 64) v = (j - 32 < 20) ? cvb1[m * 20 + (j - 32)] : 0.f;
            else             v = (j - 64 < 21) ? cvb2[m * 21 + (j - 64)] : 0.f;
            sBcv[m * 96 + j] = v;
          }
      }
      for (int j = tid; j < 112; j += PPB) {
        sBsv[j]       = (j < 100) ? svb0[idx * 100 + j] : 0.f;
        sBsv[112 + j] = (j < 100) ? svb1[idx * 100 + j] : 0.f;
        sBsv[224 + j] = (j < 100) ? svb2[idx * 100 + j] : 0.f;
      }
      if (tid < 16) sBsv[336 + tid] = (tid == 0) ? svb3[idx] : 0.f;
      for (int j = tid; j < 64; j += PPB) {
        sBhv[j]      = (j < 20) ? hb0[idx*20 + j] : (j < 40) ? db0[idx*20 + j-20]
                     : (j < 60) ? vb0[idx*20 + j-40] : 0.f;
        sBhv[64 + j] = (j < 20) ? hb1[idx*20 + j] : (j < 40) ? db1[idx*20 + j-20]
                     : (j < 60) ? vb1[idx*20 + j-40] : 0.f;
      }
      if (tid < 16)
        sBhv[128 + tid] = (tid == 0) ? hb2[idx] : (tid == 1) ? db2[idx]
                        : (tid == 2) ? vb2[idx] : 0.f;
      __syncthreads();
    }

    const float t0 = tg[i - 1], t1 = tg[i];
    const float h = t1 - t0, sh = sqrtf(h);

    // ---- svol net [t,Slog,V] -> 100 -> 100 -> 100 -> 1 (softplus) ----
    build_A0(bufA, lane, t0, Slog, V);
    zero_buf(bufB, lane); __syncthreads();
    gemm_layer<1,7,0>(sWsv + SV_L1, sBsv + 0,   bufA, bufB, nullptr,0,nullptr,nullptr,0, lane, pw);
    __syncthreads();
    zero_buf(bufA, lane); __syncthreads();
    gemm_layer<4,7,0>(sWsv + SV_L2, sBsv + 112, bufB, bufA, nullptr,0,nullptr,nullptr,0, lane, pw);
    __syncthreads();
    zero_buf(bufB, lane); __syncthreads();
    gemm_layer<4,7,0>(sWsv + SV_L3, sBsv + 224, bufA, bufB, nullptr,0,nullptr,nullptr,0, lane, pw);
    __syncthreads();
    gemm_layer<4,1,1>(sWsv + SV_L4, sBsv + 336, bufB, nullptr, sPd, 1, nullptr,nullptr,0, lane, pw);
    __syncthreads();

    // ---- fused hedge/vdrift/vvol net (block-diag 3 -> 60 -> 60 -> 3) ----
    build_A0(bufA, lane, t0, Slog, V);
    zero_buf(bufB, lane); __syncthreads();
    gemm_layer<1,4,0>(sWhv + HV_L1, sBhv + 0,  bufA, bufB, nullptr,0,nullptr,nullptr,0, lane, pw);
    __syncthreads();
    zero_buf(bufA, lane); __syncthreads();
    gemm_layer<2,4,0>(sWhv + HV_L2, sBhv + 64, bufB, bufA, nullptr,0,nullptr,nullptr,0, lane, pw);
    __syncthreads();
    gemm_layer<2,1,1>(sWhv + HV_L3, sBhv + 128, bufA, nullptr, sHvv, 3, nullptr,nullptr,0, lane, pw);
    __syncthreads();

    // ---- per-lane tamed-Euler SDE update ----
    const float zi  = z[pg * NSTEP + (i - 1)];
    const float zzi = zz[pg * NSTEP + (i - 1)];
    const float dW = sh * zi;
    const float dB = rho_t * dW + sqrtf(fmaxf(1.f - rho_t * rho_t, 0.f)) * sh * zzi;
    const float pdv   = sp_f(sPd[pl]);
    const float hedge = sHvv[pl];
    const float vdr   = sHvv[PPB + pl];
    const float vvo   = sp_f(sHvv[2 * PPB + pl]);
    const float V_new = V + vdr * h + vvo * dB;
    const float drift = rate - 0.5f * pdv * pdv;
    const float diff  = pdv * dW;
    const float drift_c = 1.f + fabsf(drift) * sh;
    const float diff_c  = 1.f + fabsf(pdv) * sh;
    const float Slog_new = Slog + drift * h / drift_c + diff / diff_c;
    const float S_new = expf(Slog_new);
    const float S_old = expf(Slog);
    const float dS = expf(-rate * t1) * S_new - expf(-rate * t0) * S_old;
    sDS[pl] = dS;
    cv_ex += hedge * dS;
    run_max = fmaxf(run_max, S_new);
    __syncthreads();

    // ---- cv nets, maturities m >= idx, accumulate cv_van += softplus(y)*dS ----
    #pragma clang loop unroll(disable)
    for (int m = idx; m < 4; ++m) {
      build_A0(bufA, lane, t0, Slog, V);
      zero_buf(bufB, lane); __syncthreads();
      gemm_layer<1,2,0>(sWcv + m*CV_MAT + CV_L1, sBcv + m*96 + 0,  bufA, bufB, nullptr,0,nullptr,nullptr,0, lane, pw);
      __syncthreads();
      zero_buf(bufA, lane); __syncthreads();
      gemm_layer<1,2,0>(sWcv + m*CV_MAT + CV_L2, sBcv + m*96 + 32, bufB, bufA, nullptr,0,nullptr,nullptr,0, lane, pw);
      __syncthreads();
      gemm_layer<1,2,2>(sWcv + m*CV_MAT + CV_L3, sBcv + m*96 + 64, bufA, nullptr, nullptr, 0, sCvv, sDS, m, lane, pw);
      __syncthreads();
    }

    // ---- commit state; maturity statistics ----
    Slog = Slog_new;
    V = fmaxf(V_new, 0.f);
    if ((i % PL) == 0) {
      const float dism = expf(-rate * t1);
      for (int k = 0; k < 21; ++k) {
        float X = dism * fmaxf(S_new - strikes[k], 0.f) - sCvv[pl * 84 + idx * 21 + k];
        float s1 = wred(X), s2 = wred(X * X);
        if (lane == 0) {
          atomicAdd(&stats[idx * 21 + k], s1);
          atomicAdd(&stats[84 + idx * 21 + k], s2);
        }
      }
    }
    __syncthreads();
  }

  // ---- terminal statistics ----
  const float disT = expf(-rate * tg[NSTEP]);
  const float Sf = expf(Slog);
  const float exo = run_max - Sf;
  const float peh = disT * exo - cv_ex;
  const float peu = disT * exo;
  float acc[7] = { disT * Sf, fmaxf(S0 - Sf, 0.f), fmaxf(Sf - S0, 0.f),
                   peh, peh * peh, peu, peu * peu };
  #pragma unroll
  for (int q = 0; q < 7; ++q) {
    float s = wred(acc[q]);
    if (lane == 0) atomicAdd(&stats[168 + q], s);
  }
}

// ---------------- finalize ----------------
__global__ void k_final(const float* __restrict__ stats, float* __restrict__ out,
                        const float* ratep, const float* tg, const float* S0p) {
  int j = threadIdx.x;
  if (j >= 176) return;
  const float N = (float)MC;
  if (j < 84) { out[j] = stats[j] / N; return; }
  if (j < 168) { int k = j - 84;
    out[j] = (stats[84 + k] - stats[k] * stats[k] / N) / (N - 1.f); return; }
  const float dis = expf(-ratep[0] * tg[NSTEP]);
  const float S0 = S0p[0];
  float r = 0.f;
  switch (j) {
    case 168: r = stats[173] / N; break;                                      // pe_u
    case 169: r = stats[171] / N; break;                                      // pe_h
    case 170: r = (stats[174] - stats[173]*stats[173]/N) / (N - 1.f); break;  // pe_var_u
    case 171: r = (stats[172] - stats[171]*stats[171]/N) / (N - 1.f); break;  // pe_var_h
    case 172: r = stats[168] / N / S0; break;                                 // martingale
    case 173: r = dis * stats[169] / N; break;                                // put_atm
    case 174: r = dis * stats[170] / N; break;                                // call_atm
    case 175: r = dis * (stats[170] - stats[169]) / N - S0 + dis * S0; break; // parity err
  }
  out[j] = r;
}

// ---------------- launch ----------------
extern "C" void kernel_launch(void* const* d_in, const int* in_sizes, int n_in,
                              void* d_out, int out_size, void* d_ws, size_t ws_size,
                              hipStream_t stream) {
  const float* S0p  = (const float*)d_in[0];
  const float* ratep= (const float*)d_in[1];
  const float* z    = (const float*)d_in[2];
  const float* zz   = (const float*)d_in[3];
  const float* tg   = (const float*)d_in[4];
  const float* strk = (const float*)d_in[5];
  const float* Wsv0 = (const float*)d_in[6],  *bsv0 = (const float*)d_in[7];
  const float* Wsv1 = (const float*)d_in[8],  *bsv1 = (const float*)d_in[9];
  const float* Wsv2 = (const float*)d_in[10], *bsv2 = (const float*)d_in[11];
  const float* Wsv3 = (const float*)d_in[12], *bsv3 = (const float*)d_in[13];
  const float* Wcv0 = (const float*)d_in[14], *bcv0 = (const float*)d_in[15];
  const float* Wcv1 = (const float*)d_in[16], *bcv1 = (const float*)d_in[17];
  const float* Wcv2 = (const float*)d_in[18], *bcv2 = (const float*)d_in[19];
  const float* Wvd0 = (const float*)d_in[20], *bvd0 = (const float*)d_in[21];
  const float* Wvd1 = (const float*)d_in[22], *bvd1 = (const float*)d_in[23];
  const float* Wvd2 = (const float*)d_in[24], *bvd2 = (const float*)d_in[25];
  const float* Wvv0 = (const float*)d_in[26], *bvv0 = (const float*)d_in[27];
  const float* Wvv1 = (const float*)d_in[28], *bvv1 = (const float*)d_in[29];
  const float* Wvv2 = (const float*)d_in[30], *bvv2 = (const float*)d_in[31];
  const float* Whe0 = (const float*)d_in[32], *bhe0 = (const float*)d_in[33];
  const float* Whe1 = (const float*)d_in[34], *bhe1 = (const float*)d_in[35];
  const float* Whe2 = (const float*)d_in[36], *bhe2 = (const float*)d_in[37];
  const float* v0p  = (const float*)d_in[38];
  const float* rhop = (const float*)d_in[39];

  char* ws = (char*)d_ws;
  float*     stats = (float*)ws;
  _Float16*  wsv   = (_Float16*)(ws + WS_SV);
  _Float16*  whv   = (_Float16*)(ws + WS_HV);
  _Float16*  wcv   = (_Float16*)(ws + WS_CV);

  hipMemsetAsync(stats, 0, 2048, stream);

  auto PB = [](int tiles) { return (tiles * 512 + 255) / 256; };
  // svol
  k_pack<<<dim3(PB(7), 4),  256, 0, stream>>>(Wsv0, 300,   wsv + SV_L1, SV_MAT, 3,   100, 1, 7);
  k_pack<<<dim3(PB(28),4),  256, 0, stream>>>(Wsv1, 10000, wsv + SV_L2, SV_MAT, 100, 100, 4, 7);
  k_pack<<<dim3(PB(28),4),  256, 0, stream>>>(Wsv2, 10000, wsv + SV_L3, SV_MAT, 100, 100, 4, 7);
  k_pack<<<dim3(PB(4), 4),  256, 0, stream>>>(Wsv3, 100,   wsv + SV_L4, SV_MAT, 100, 1,   4, 1);
  // cv
  k_pack<<<dim3(PB(2), 4),  256, 0, stream>>>(Wcv0, 40,  wcv + CV_L1, CV_MAT, 2,  20, 1, 2);
  k_pack<<<dim3(PB(2), 4),  256, 0, stream>>>(Wcv1, 400, wcv + CV_L2, CV_MAT, 20, 20, 1, 2);
  k_pack<<<dim3(PB(2), 4),  256, 0, stream>>>(Wcv2, 420, wcv + CV_L3, CV_MAT, 20, 21, 1, 2);
  // fused hedge/vdrift/vvol (block-diagonal)
  k_pack3<<<dim3(PB(4), 4), 256, 0, stream>>>(
      Blk{Whe0, 40, 0, 2, 0, 20, 20}, Blk{Wvd0, 20, 2, 3, 20, 40, 20},
      Blk{Wvv0, 20, 2, 3, 40, 60, 20}, whv + HV_L1, HV_MAT, 1, 4);
  k_pack3<<<dim3(PB(8), 4), 256, 0, stream>>>(
      Blk{Whe1, 400, 0, 20, 0, 20, 20}, Blk{Wvd1, 400, 20, 40, 20, 40, 20},
      Blk{Wvv1, 400, 40, 60, 40, 60, 20}, whv + HV_L2, HV_MAT, 2, 4);
  k_pack3<<<dim3(PB(2), 4), 256, 0, stream>>>(
      Blk{Whe2, 20, 0, 20, 0, 1, 1}, Blk{Wvd2, 20, 20, 40, 1, 2, 1},
      Blk{Wvv2, 20, 40, 60, 2, 3, 1}, whv + HV_L3, HV_MAT, 2, 1);

  k_sim<<<MC / PPB, PPB, 0, stream>>>(
      S0p, ratep, z, zz, tg, strk,
      bsv0, bsv1, bsv2, bsv3,
      bcv0, bcv1, bcv2,
      bhe0, bhe1, bhe2,
      bvd0, bvd1, bvd2,
      bvv0, bvv1, bvv2,
      v0p, rhop, wsv, whv, wcv, stats);

  k_final<<<1, 256, 0, stream>>>(stats, (float*)d_out, ratep, tg, S0p);
}
